// SAGE_64845416235755
// MI455X (gfx1250) — matrix operand order repaired and checked by reading, not compiled, not run on hardware
//
#include <hip/hip_runtime.h>
#include <hip/hip_bf16.h>

// ---------------------------------------------------------------------------
// GraphSAGE 3-layer forward for MI455X (gfx1250), wave32 + WMMA bf16.
//   layer: out = [h | mean_agg(h)] @ [W_self; W_neigh] + b  (K = 256 fused)
//   between layers: relu(out) * dropout_mask
// Aggregation: fp32 atomics into L2-resident accumulator (whole working set
// ~100MB < 192MB L2, so scatter/gather runs at L2 speed, not HBM).
// Dense math: v_wmma_f32_16x16x32_bf16 (8 k-steps x 4 col-tiles per wave).
// ---------------------------------------------------------------------------

#define NN 50000
#define NE 800000

typedef __bf16 bf16_t;
typedef bf16_t bf16x8 __attribute__((ext_vector_type(8)));
typedef bf16_t v16bf  __attribute__((ext_vector_type(16)));
typedef float  v8f    __attribute__((ext_vector_type(8)));

union Frag16 { v16bf v; bf16x8 h[2]; };

// ---------------- degree / inverse degree ----------------------------------
__global__ void deg_kernel(const int* __restrict__ dst, float* __restrict__ deg, int E) {
    int t = blockIdx.x * blockDim.x + threadIdx.x;
    if (t < E) atomicAdd(&deg[dst[t]], 1.0f);
}

__global__ void invdeg_kernel(float* __restrict__ deg, int n) {
    int t = blockIdx.x * blockDim.x + threadIdx.x;
    if (t < n) deg[t] = 1.0f / fmaxf(deg[t], 1.0f);
}

// ---------------- weight concat + fp32 -> bf16 -----------------------------
// wcat[k][c] = (k<128 ? wself[k][c] : wneigh[k-128][c]), k in [0,256)
__global__ void wcat_kernel(const float* __restrict__ wself, const float* __restrict__ wneigh,
                            bf16_t* __restrict__ wcat, int ncols, int shift) {
    int t = blockIdx.x * blockDim.x + threadIdx.x;
    if (t >= 256 * ncols) return;
    int k = t >> shift;
    int c = t & (ncols - 1);
    float v = (k < 128) ? wself[k * ncols + c] : wneigh[(k - 128) * ncols + c];
    wcat[t] = (bf16_t)v;
}

// ---------------- edge scatter-add: agg[dst] += h[src] ---------------------
// one wave32 per edge; lane j handles floats [4j, 4j+4).
// src/dst indices are wave-uniform -> force to SGPRs (scalar loads).
__global__ void scatter_kernel(const float* __restrict__ h, const int* __restrict__ src,
                               const int* __restrict__ dst, float* __restrict__ agg, int E) {
    int t = blockIdx.x * blockDim.x + threadIdx.x;
    int e = __builtin_amdgcn_readfirstlane(t >> 5);   // uniform per wave32
    if (e >= E) return;
    int j = t & 31;
    int s = src[e];                                    // scalar load
    int d = dst[e];                                    // scalar load
    float4 v = reinterpret_cast<const float4*>(h + (size_t)s * 128)[j];
    float* a = agg + (size_t)d * 128 + j * 4;
    atomicAdd(a + 0, v.x);
    atomicAdd(a + 1, v.y);
    atomicAdd(a + 2, v.z);
    atomicAdd(a + 3, v.w);
}

// ---------------- build bf16 Hcat[N][256] = [h | agg*inv_deg] --------------
__global__ void prep_kernel(const float* __restrict__ h, const float* __restrict__ agg,
                            const float* __restrict__ invd, bf16_t* __restrict__ hcat, int n) {
    int t = blockIdx.x * blockDim.x + threadIdx.x;
    int node = t >> 6;
    if (node >= n) return;
    int j = t & 63;                                   // 64 float2 chunks per node
    float id = invd[node];
    float2 hv = reinterpret_cast<const float2*>(h + (size_t)node * 128)[j];
    float2 av = reinterpret_cast<const float2*>(agg + (size_t)node * 128)[j];
    union { bf16_t b[2]; unsigned u; } p;
    unsigned* row = reinterpret_cast<unsigned*>(hcat + (size_t)node * 256);
    p.b[0] = (bf16_t)hv.x; p.b[1] = (bf16_t)hv.y;
    row[j] = p.u;                                     // self half: cols [0,128)
    p.b[0] = (bf16_t)(av.x * id); p.b[1] = (bf16_t)(av.y * id);
    row[64 + j] = p.u;                                // neigh half: cols [128,256)
}

// ---------------- WMMA GEMM: out[N][ncols] = Hcat @ Wcat (+bias, relu*mask) -
// block = 256 threads = 8 waves; wave w -> rows [blk*128 + 16w, +16), 64 cols.
__launch_bounds__(256)
__global__ void sage_gemm_kernel(const bf16_t* __restrict__ hcat,
                                 const bf16_t* __restrict__ wcat,
                                 const float* __restrict__ bias,
                                 const float* __restrict__ mask,
                                 float* __restrict__ out,
                                 int nrows, int ncols, int do_relu) {
    // LDS: weight tile transposed [local_col][k], padded stride 264 (528B = 33*16B)
    __shared__ bf16_t ldsB[64 * 264];
    const int tid = threadIdx.x;
    const int colBase = blockIdx.y * 64;

    // cooperative 256x64 weight tile load (coalesced global reads)
    for (int idx = tid; idx < 256 * 64; idx += 256) {
        int k = idx >> 6;
        int c = idx & 63;
        ldsB[c * 264 + k] = wcat[k * ncols + colBase + c];
    }
    __syncthreads();

    const int wave = tid >> 5;
    const int lane = tid & 31;
    const int half = lane >> 4;     // 0: lanes 0-15, 1: lanes 16-31
    const int l16  = lane & 15;
    const int rBase = blockIdx.x * 128 + wave * 16;

    int rowA = rBase + l16;                  // A row this lane supplies
    if (rowA > nrows - 1) rowA = nrows - 1;  // clamp; stores are guarded below

    v8f acc[4] = {};

    for (int k0 = 0; k0 < 256; k0 += 32) {
        // A fragment (16x32 bf16): elems 0-7 <- K=k0+8*half+0..7, elems 8-15 <- +16
        Frag16 fa;
        const bf16x8* ap =
            reinterpret_cast<const bf16x8*>(hcat + (size_t)rowA * 256 + k0 + 8 * half);
        fa.h[0] = ap[0];
        fa.h[1] = ap[2];

        // load ALL four B fragments first (distinct regs), then issue the four
        // WMMAs -> LDS waits overlap with matrix issue instead of serializing
        const int kb = k0 + 8 * half;        // B: elems 0-7 <- K=k0+8*half+0..7, elems 8-15 <- +16
        Frag16 fb[4];
#pragma unroll
        for (int c = 0; c < 4; ++c) {
            const bf16x8* bp =
                reinterpret_cast<const bf16x8*>(&ldsB[(c * 16 + l16) * 264 + kb]);
            fb[c].h[0] = bp[0];
            fb[c].h[1] = bp[2];
        }
#pragma unroll
        for (int c = 0; c < 4; ++c) {
            acc[c] = __builtin_amdgcn_wmma_f32_16x16x32_bf16(
                false, fa.v, false, fb[c].v, (short)0, acc[c], false, false);
        }
    }

    // epilogue: bias (+ relu * dropout-mask), fp32 store
#pragma unroll
    for (int c = 0; c < 4; ++c) {
        int col = colBase + c * 16 + l16;
        float bv = bias[col];
#pragma unroll
        for (int r = 0; r < 8; ++r) {
            int row = rBase + 8 * half + r;  // C/D layout: M = r + 8*half
            if (row < nrows) {
                float v = acc[c][r] + bv;
                if (do_relu) v = fmaxf(v, 0.0f) * mask[(size_t)row * 128 + col];
                out[(size_t)row * ncols + col] = v;
            }
        }
    }
}

// ---------------------------------------------------------------------------
extern "C" void kernel_launch(void* const* d_in, const int* in_sizes, int n_in,
                              void* d_out, int out_size, void* d_ws, size_t ws_size,
                              hipStream_t stream) {
    const float* x        = (const float*)d_in[0];
    const int*   src      = (const int*)  d_in[1];
    const int*   dst      = (const int*)  d_in[2];
    const float* Wself1   = (const float*)d_in[3];
    const float* Wneigh1  = (const float*)d_in[4];
    const float* b1       = (const float*)d_in[5];
    const float* Wself2   = (const float*)d_in[6];
    const float* Wneigh2  = (const float*)d_in[7];
    const float* b2       = (const float*)d_in[8];
    const float* Wself3   = (const float*)d_in[9];
    const float* Wneigh3  = (const float*)d_in[10];
    const float* b3       = (const float*)d_in[11];
    const float* mask1    = (const float*)d_in[12];
    const float* mask2    = (const float*)d_in[13];
    float* outp = (float*)d_out;

    // workspace carve-up (256B aligned)
    char* ws = (char*)d_ws;
    size_t off = 0;
    auto carve = [&](size_t bytes) -> void* {
        void* p = ws + off;
        off += (bytes + 255) & ~(size_t)255;
        return p;
    };
    float*  invdeg = (float*)carve((size_t)NN * 4);
    float*  hA     = (float*)carve((size_t)NN * 128 * 4);
    float*  hB     = (float*)carve((size_t)NN * 128 * 4);
    float*  agg    = (float*)carve((size_t)NN * 128 * 4);
    bf16_t* hcat   = (bf16_t*)carve((size_t)NN * 256 * 2);
    bf16_t* wc1    = (bf16_t*)carve(256 * 128 * 2);
    bf16_t* wc2    = (bf16_t*)carve(256 * 128 * 2);
    bf16_t* wc3    = (bf16_t*)carve(256 * 64 * 2);

    const int TPB = 256;
    const int gridE   = (NE + TPB - 1) / TPB;
    const int gridN   = (NN + TPB - 1) / TPB;
    const int gridE32 = (NE * 32 + TPB - 1) / TPB;          // 100000 blocks
    const int gridN64 = (NN * 64 + TPB - 1) / TPB;
    const int gRows   = (NN + 127) / 128;                   // 391

    // degrees (shared by all layers)
    hipMemsetAsync(invdeg, 0, (size_t)NN * 4, stream);
    deg_kernel<<<gridE, TPB, 0, stream>>>(dst, invdeg, NE);
    invdeg_kernel<<<gridN, TPB, 0, stream>>>(invdeg, NN);

    // bf16 concatenated weights
    wcat_kernel<<<(256 * 128) / TPB, TPB, 0, stream>>>(Wself1, Wneigh1, wc1, 128, 7);
    wcat_kernel<<<(256 * 128) / TPB, TPB, 0, stream>>>(Wself2, Wneigh2, wc2, 128, 7);
    wcat_kernel<<<(256 * 64)  / TPB, TPB, 0, stream>>>(Wself3, Wneigh3, wc3, 64, 6);

    // -------- layer 1 --------
    hipMemsetAsync(agg, 0, (size_t)NN * 128 * 4, stream);
    scatter_kernel<<<gridE32, TPB, 0, stream>>>(x, src, dst, agg, NE);
    prep_kernel<<<gridN64, TPB, 0, stream>>>(x, agg, invdeg, hcat, NN);
    sage_gemm_kernel<<<dim3(gRows, 2), TPB, 0, stream>>>(hcat, wc1, b1, mask1, hA, NN, 128, 1);

    // -------- layer 2 --------
    hipMemsetAsync(agg, 0, (size_t)NN * 128 * 4, stream);
    scatter_kernel<<<gridE32, TPB, 0, stream>>>(hA, src, dst, agg, NE);
    prep_kernel<<<gridN64, TPB, 0, stream>>>(hA, agg, invdeg, hcat, NN);
    sage_gemm_kernel<<<dim3(gRows, 2), TPB, 0, stream>>>(hcat, wc2, b2, mask2, hB, NN, 128, 1);

    // -------- layer 3 (no relu/mask, fp32 out [N,64]) --------
    hipMemsetAsync(agg, 0, (size_t)NN * 128 * 4, stream);
    scatter_kernel<<<gridE32, TPB, 0, stream>>>(hB, src, dst, agg, NE);
    prep_kernel<<<gridN64, TPB, 0, stream>>>(hB, agg, invdeg, hcat, NN);
    sage_gemm_kernel<<<dim3(gRows, 1), TPB, 0, stream>>>(hcat, wc3, b3, b3 /*unused*/, outp, NN, 64, 0);
}
